// PredictOrderingModule_25340307046802
// MI455X (gfx1250) — matrix operand order repaired and checked by reading, not compiled, not run on hardware
//
#include <hip/hip_runtime.h>
#include <hip/hip_bf16.h>

// ---------------- CDNA5 WMMA types ----------------
typedef __attribute__((ext_vector_type(16))) __bf16 v16bf;
typedef __attribute__((ext_vector_type(8)))  float  v8f;

union Frag { uint4 u[2]; v16bf v; };   // 16 bf16 = two 16B chunks

#define NB   16
#define NPD  512
#define NOD  512
#define TT   2048
#define NF   8
#define HS   256
#define TR   2030   // T - (DELAY + SKIP*(NF-1)) = 2048-18

__device__ __forceinline__ unsigned short f2bf(float f) {
    unsigned int u = __float_as_uint(f);
    unsigned int r = u + 0x7FFFu + ((u >> 16) & 1u);   // round to nearest even
    return (unsigned short)(r >> 16);
}

__device__ __forceinline__ v8f vzero8() {
    v8f z = {0.f,0.f,0.f,0.f,0.f,0.f,0.f,0.f};
    return z;
}

// ---------------- Kernel 1: convert weights to bf16 ----------------
__global__ void k_convert(const float* __restrict__ tri, const float* __restrict__ w_o,
                          const float* __restrict__ w_p,
                          unsigned short* __restrict__ triB,
                          unsigned short* __restrict__ woB,
                          unsigned short* __restrict__ wpB) {
    int i = blockIdx.x * 256 + threadIdx.x;          // 8192 blocks -> 2,097,152 threads
    if (i < NF * NOD * NPD) triB[i] = f2bf(tri[i]);
    if (i < HS * NOD) { woB[i] = f2bf(w_o[i]); wpB[i] = f2bf(w_p[i]); }
}

// ---------------- Kernel 2: projection GEMM (o_proj / p_proj) ----------------
// out[b,h,t] = sum_p wB[h,p]*src[b,p,min(t+shift,2047)] + bias[h]   for t<2048
// M=256 (16 tiles), N=64 per block, K=512. 8 waves: wave w -> rows [w*32,w*32+32).
// B tile staged TRANSPOSED: row t (64), 32 K bf16 -> 4 uint4, stride 5 uint4.
__global__ void k_proj(const float* __restrict__ src,
                       const unsigned short* __restrict__ wB,
                       const float* __restrict__ bias,
                       float* __restrict__ out, int shift) {
    __shared__ uint4 Blds4[64 * 5];                        // 5,120 B
    unsigned short* BldsU = (unsigned short*)Blds4;        // row stride 40 ushorts

    const int b = blockIdx.y, t0 = blockIdx.x * 64;
    const int tid = threadIdx.x, lane = tid & 31, w = tid >> 5;
    const int half = lane >> 4, lr = lane & 15;

    v8f acc[2][4];
    for (int a = 0; a < 2; a++) for (int c = 0; c < 4; c++) acc[a][c] = vzero8();

    const float* srcb = src + (size_t)b * NPD * TT;

    for (int p0 = 0; p0 < NPD; p0 += 32) {
        for (int e = tid; e < 32 * 64; e += 256) {
            int pp = e >> 6, tt = e & 63;                  // coalesced over t
            int tg = t0 + tt + shift; if (tg > TT - 1) tg = TT - 1;
            BldsU[tt * 40 + pp] = f2bf(srcb[(size_t)(p0 + pp) * TT + tg]);
        }
        __syncthreads();

        Frag af[2];
        #pragma unroll
        for (int mt = 0; mt < 2; mt++) {
            int h = w * 32 + mt * 16 + lr;
            const uint4* ap = (const uint4*)(wB + (size_t)h * NPD + p0 + half * 8);
            af[mt].u[0] = ap[0];
            af[mt].u[1] = ap[2];           // +16 halfwords
        }
        #pragma unroll
        for (int tt4 = 0; tt4 < 4; tt4++) {
            int tcol = tt4 * 16 + lr;
            Frag bf;
            bf.u[0] = Blds4[tcol * 5 + half];              // K = half*8 + 0..7
            bf.u[1] = Blds4[tcol * 5 + half + 2];          // K = 16 + half*8 + 0..7
            #pragma unroll
            for (int mt = 0; mt < 2; mt++)
                acc[mt][tt4] = __builtin_amdgcn_wmma_f32_16x16x32_bf16(
                    false, af[mt].v, false, bf.v, (short)0, acc[mt][tt4], false, false);
        }
        __syncthreads();
    }

    #pragma unroll
    for (int mt = 0; mt < 2; mt++)
        #pragma unroll
        for (int tt4 = 0; tt4 < 4; tt4++)
            #pragma unroll
            for (int r = 0; r < 8; r++) {
                int h = w * 32 + mt * 16 + r + half * 8;
                int t = t0 + tt4 * 16 + lr;
                out[((size_t)b * HS + h) * TT + t] = acc[mt][tt4][r] + bias[h];
            }
}

// ---------------- Kernel 3: fused pp-GEMM + j-shift contraction -> inner ----------------
// Block = (tb, k, b). PP[m,t] = sum_p tri[k,m,p]*pred[b,p,t0+t] for m=0..511, t tile 16,
// then inner[b,j,k,t] = sum_m PP[m,t]*to_order[b,m,4+t+2j].
// Full pred tile (16 t x 512 K) staged transposed ONCE -> barrier-free GEMM loop.
__global__ void k_inner(const float* __restrict__ predictor,
                        const float* __restrict__ to_order,
                        const unsigned short* __restrict__ triB,
                        float* __restrict__ inner) {
    __shared__ __align__(16) char smem[NOD * 17 * 4 + 16 * 65 * 16];   // 51,456 B
    float* PP  = (float*)smem;                                // 512*17 f32 (34,816 B)
    uint4* Bt4 = (uint4*)(smem + NOD * 17 * 4);               // 16 rows x 65 uint4
    unsigned short* BtU = (unsigned short*)Bt4;               // row stride 520 ushorts
    float* TOO = (float*)(smem + NOD * 17 * 4);               // aliases Bt4 (dead then)

    const int tb = blockIdx.x, k = blockIdx.y, b = blockIdx.z;
    const int t0 = tb * 16;
    const int tid = threadIdx.x, lane = tid & 31, w = tid >> 5;
    const int half = lane >> 4, lr = lane & 15;

    const float* predb = predictor + (size_t)b * NPD * TT;
    const unsigned short* trik = triB + (size_t)k * NOD * NPD;

    // stage full B tile transposed: BtU[t][p]
    for (int e = tid; e < NPD * 16; e += 256) {
        int p = e >> 4, t = e & 15;
        BtU[t * 520 + p] = f2bf(predb[(size_t)p * TT + t0 + t]);
    }
    __syncthreads();

    v8f acc[4];
    for (int c = 0; c < 4; c++) acc[c] = vzero8();

    for (int p0 = 0; p0 < NPD; p0 += 32) {
        Frag bf;
        bf.u[0] = Bt4[lr * 65 + (p0 >> 3) + half];
        bf.u[1] = Bt4[lr * 65 + (p0 >> 3) + half + 2];
        #pragma unroll
        for (int mt = 0; mt < 4; mt++) {
            int m = w * 64 + mt * 16 + lr;
            Frag af;
            const uint4* ap = (const uint4*)(trik + (size_t)m * NPD + p0 + half * 8);
            af.u[0] = ap[0]; af.u[1] = ap[2];
            acc[mt] = __builtin_amdgcn_wmma_f32_16x16x32_bf16(
                false, af.v, false, bf.v, (short)0, acc[mt], false, false);
        }
    }

    // park PP in LDS (separate region; no barrier needed vs Bt reads)
    #pragma unroll
    for (int mt = 0; mt < 4; mt++)
        #pragma unroll
        for (int r = 0; r < 8; r++) {
            int m = w * 64 + mt * 16 + r + half * 8;
            PP[m * 17 + lr] = acc[mt][r];
        }

    // phase 2: 256 threads, one lane-pair per (j,t) dot, m split between the pair
    const float* toob = to_order + (size_t)b * NOD * TT;
    const int task = tid >> 1, pair = tid & 1;
    const int j = task >> 4, t = task & 15;
    const bool act = (t0 + t < TR);
    float s = 0.0f;

    for (int m0 = 0; m0 < NOD; m0 += 128) {
        __syncthreads();   // Bt/TOO region free; PP writes visible
        for (int e = tid; e < 128 * 30; e += 256) {
            int m = e / 30, tt = e % 30;
            int srci = 4 + t0 + tt; if (srci > TT - 1) srci = TT - 1;
            TOO[m * 31 + tt] = toob[(size_t)(m0 + m) * TT + srci];
        }
        __syncthreads();
        if (act) {
            int mlo = pair * 64;
            #pragma unroll 4
            for (int m = mlo; m < mlo + 64; m++)
                s += PP[(m0 + m) * 17 + t] * TOO[m * 31 + t + 2 * j];
        }
    }
    s += __shfl_xor(s, 1, 32);
    if (act && pair == 0)
        inner[(((size_t)b * NF + j) * NF + k) * TT + t0 + t] = s;
}

// ---------------- Kernel 4: batchnorm stats (deterministic, one block per h) ----------------
__global__ void k_stats(const float* __restrict__ o_proj, const float* __restrict__ p_proj,
                        float* __restrict__ stats) {
    const int h = blockIdx.x, tid = threadIdx.x;
    float s = 0.f, s2 = 0.f;
    for (int b = 0; b < NB; b++) {
        const float* orow = o_proj + ((size_t)b * HS + h) * TT;
        const float* prow = p_proj + ((size_t)b * HS + h) * TT;
        for (int t = tid; t < TR; t += 256) {
            float pv = prow[t];
            #pragma unroll
            for (int j = 0; j < NF; j++) {
                float v = orow[t + 2 * j] + pv;
                v = v > 0.f ? v : 0.f;
                s += v; s2 += v * v;
            }
        }
    }
    __shared__ float rs[256], rs2[256];
    rs[tid] = s; rs2[tid] = s2; __syncthreads();
    for (int off = 128; off > 0; off >>= 1) {
        if (tid < off) { rs[tid] += rs[tid + off]; rs2[tid] += rs2[tid + off]; }
        __syncthreads();
    }
    if (tid == 0) { stats[h] = rs[0]; stats[HS + h] = rs2[0]; }
}

// ---------------- Kernel 5: fold BN into head weights ----------------
__global__ void k_finalize(const float* __restrict__ stats,
                           const float* __restrict__ gamma, const float* __restrict__ beta,
                           const float* __restrict__ w_out, const float* __restrict__ b_out,
                           unsigned short* __restrict__ woutaB, float* __restrict__ kbias) {
    const int h = threadIdx.x;
    __shared__ float c_sh[HS];
    const float cnt = (float)NB * NF * TR;
    float mean = stats[h] / cnt;
    float var  = stats[HS + h] / cnt - mean * mean;
    float a = gamma[h] * rsqrtf(var + 1e-5f);
    float c = beta[h] - a * mean;
    c_sh[h] = c;
    for (int k = 0; k < NF; k++)  woutaB[k * HS + h] = f2bf(w_out[k * HS + h] * a);
    for (int k = NF; k < 16; k++) woutaB[k * HS + h] = 0;   // zero-pad rows 8..15
    __syncthreads();
    if (h < NF) {
        float kb = b_out[h];
        for (int hh = 0; hh < HS; hh++) kb += w_out[h * HS + hh] * c_sh[hh];
        kbias[h] = kb;
    }
}

// ---------------- Kernel 6: head GEMM + log_softmax + partial reduce ----------------
// Block = (tblock64, j, b). H[t][h] = relu(o_proj[b,h,t+2j]+p_proj[b,h,t]) in LDS (bf16,
// transposed), head = woutaB(16x256) x H -> +inner+kbias -> log_softmax over k -> pick k=j.
__global__ void k_head(const float* __restrict__ o_proj, const float* __restrict__ p_proj,
                       const unsigned short* __restrict__ woutaB,
                       const float* __restrict__ kbias,
                       const float* __restrict__ inner, float* __restrict__ partial) {
    __shared__ uint4 Hv4[64 * 33];                       // 33,792 B
    unsigned short* HldsU = (unsigned short*)Hv4;        // row stride 264 ushorts
    const int tbk = blockIdx.x, j = blockIdx.y, b = blockIdx.z;
    const int t0 = tbk * 64;
    const int tid = threadIdx.x, lane = tid & 31, w = tid >> 5;

    const float* ob = o_proj + (size_t)b * HS * TT;
    const float* pb = p_proj + (size_t)b * HS * TT;
    for (int e = tid; e < HS * 64; e += 256) {
        int h = e >> 6, t = e & 63;
        int tg = t0 + t;
        int io = tg + 2 * j; if (io > TT - 1) io = TT - 1;
        float v = ob[(size_t)h * TT + io] + pb[(size_t)h * TT + tg];
        v = v > 0.f ? v : 0.f;
        HldsU[t * 264 + h] = f2bf(v);
    }
    __syncthreads();

    if (w == 0) {
        const int half = lane >> 4, lr = lane & 15;
        v8f acc[4];
        for (int c = 0; c < 4; c++) acc[c] = vzero8();

        for (int k0 = 0; k0 < HS; k0 += 32) {
            Frag af;
            const uint4* ap = (const uint4*)(woutaB + (size_t)lr * HS + k0 + half * 8);
            af.u[0] = ap[0]; af.u[1] = ap[2];
            #pragma unroll
            for (int tt4 = 0; tt4 < 4; tt4++) {
                int tcol = tt4 * 16 + lr;
                Frag bf;
                bf.u[0] = Hv4[tcol * 33 + (k0 >> 3) + half];
                bf.u[1] = Hv4[tcol * 33 + (k0 >> 3) + half + 2];
                acc[tt4] = __builtin_amdgcn_wmma_f32_16x16x32_bf16(
                    false, af.v, false, bf.v, (short)0, acc[tt4], false, false);
            }
        }

        float local = 0.0f;
        if (half == 0) {   // lanes 0..15 hold k=0..7 in acc elements r=0..7, column t = lr
            #pragma unroll
            for (int tt4 = 0; tt4 < 4; tt4++) {
                int t = t0 + tt4 * 16 + lr;
                if (t < TR) {
                    const float* ib = inner + (((size_t)b * NF + j) * NF) * TT + t;
                    float vals[8], m = -1e30f;
                    #pragma unroll
                    for (int k = 0; k < NF; k++) {
                        vals[k] = acc[tt4][k] + ib[(size_t)k * TT] + kbias[k];
                        m = fmaxf(m, vals[k]);
                    }
                    float se = 0.f;
                    #pragma unroll
                    for (int k = 0; k < NF; k++) se += expf(vals[k] - m);
                    local += vals[j] - m - logf(se);
                }
            }
        }
        for (int off = 8; off > 0; off >>= 1) local += __shfl_down(local, off, 16);
        if (lane == 0) partial[((size_t)b * NF + j) * 32 + tbk] = local;
    }
}

// ---------------- Kernel 7: final deterministic reduce ----------------
__global__ void k_out(const float* __restrict__ partial, float* __restrict__ out) {
    int j = threadIdx.x;
    if (j < NF) {
        float s = 0.f;
        for (int b = 0; b < NB; b++)
            for (int tb = 0; tb < 32; tb++)
                s += partial[((size_t)b * NF + j) * 32 + tb];
        out[j] = s * (1.0f / ((float)TR * (float)NB));
    }
}

// ---------------- host launcher ----------------
extern "C" void kernel_launch(void* const* d_in, const int* in_sizes, int n_in,
                              void* d_out, int out_size, void* d_ws, size_t ws_size,
                              hipStream_t stream) {
    const float* predictor = (const float*)d_in[0];
    const float* to_order  = (const float*)d_in[1];
    const float* trilinear = (const float*)d_in[2];
    const float* w_o       = (const float*)d_in[3];
    const float* b_o       = (const float*)d_in[4];
    const float* w_p       = (const float*)d_in[5];
    const float* b_p       = (const float*)d_in[6];
    const float* bn_gamma  = (const float*)d_in[7];
    const float* bn_beta   = (const float*)d_in[8];
    const float* w_out     = (const float*)d_in[9];
    const float* b_out     = (const float*)d_in[10];
    float* out = (float*)d_out;

    char* ws = (char*)d_ws;
    size_t off = 0;
    unsigned short* triB   = (unsigned short*)(ws + off); off += (size_t)NF * NOD * NPD * 2;   // 4 MB
    unsigned short* woB    = (unsigned short*)(ws + off); off += (size_t)HS * NOD * 2;          // 256 KB
    unsigned short* wpB    = (unsigned short*)(ws + off); off += (size_t)HS * NPD * 2;          // 256 KB
    unsigned short* woutaB = (unsigned short*)(ws + off); off += (size_t)16 * HS * 2;           // 8 KB
    off = (off + 255) & ~(size_t)255;
    float* stats   = (float*)(ws + off); off += (size_t)2 * HS * 4;
    off = (off + 255) & ~(size_t)255;
    float* kbias   = (float*)(ws + off); off += 256;
    float* partial = (float*)(ws + off); off += (size_t)NB * NF * 32 * 4;
    off = (off + 255) & ~(size_t)255;
    float* o_proj  = (float*)(ws + off); off += (size_t)NB * HS * TT * 4;   // 32 MB
    float* p_proj  = (float*)(ws + off); off += (size_t)NB * HS * TT * 4;   // 32 MB
    float* inner   = (float*)(ws + off); off += (size_t)NB * NF * NF * TT * 4; // 8 MB

    k_convert<<<8192, 256, 0, stream>>>(trilinear, w_o, w_p, triB, woB, wpB);
    k_proj<<<dim3(32, NB), 256, 0, stream>>>(to_order,  woB, b_o, o_proj, 4);
    k_proj<<<dim3(32, NB), 256, 0, stream>>>(predictor, wpB, b_p, p_proj, 0);
    k_inner<<<dim3(127, NF, NB), 256, 0, stream>>>(predictor, to_order, triB, inner);
    k_stats<<<HS, 256, 0, stream>>>(o_proj, p_proj, stats);
    k_finalize<<<1, HS, 0, stream>>>(stats, bn_gamma, bn_beta, w_out, b_out, woutaB, kbias);
    k_head<<<dim3(32, NF, NB), 256, 0, stream>>>(o_proj, p_proj, woutaB, kbias, inner, partial);
    k_out<<<1, 32, 0, stream>>>(partial, out);
}
